// MultiHeadAttention_23562190586539
// MI455X (gfx1250) — compile-verified, matrix operand order checked by reading, not run on hardware
//
#include <hip/hip_runtime.h>
#include <hip/hip_bf16.h>
#include <stdint.h>

// ---------------------------------------------------------------------------
// Types for CDNA5 WMMA
// ---------------------------------------------------------------------------
typedef __attribute__((ext_vector_type(16))) __bf16 v16bf;
typedef __attribute__((ext_vector_type(8)))  float  v8f;

union Frag16 {
    v16bf v;
    uint4 q[2];
    unsigned int u[8];
};

#define D_MODEL 768
#define N_HEADS 12
#define D_K     64
#define SEQ     2048
#define BATCH   2
#define ROWS    (BATCH * SEQ)      // 4096
#define LOG2E   1.442695040888963f

// fp32 -> bf16 (round to nearest even), stored as raw u16
__device__ __forceinline__ unsigned short f2bf(float f) {
    unsigned int u = __float_as_uint(f);
    u += 0x7fffu + ((u >> 16) & 1u);
    return (unsigned short)(u >> 16);
}

// Load one 16-bit A/B WMMA fragment (ISA 7.12.2 layout) from a row whose
// K elements are contiguous: dwords 0..3 cover K = kbase+0..7,
// dwords 4..7 cover K = 16+kbase+0..7  ->  two aligned b128 loads.
__device__ __forceinline__ void load_frag(Frag16& f, const unsigned short* rowp,
                                          int kbase) {
    f.q[0] = *(const uint4*)(rowp + kbase);
    f.q[1] = *(const uint4*)(rowp + 16 + kbase);
}

// CDNA5 async global->LDS copy (ASYNCcnt-tracked, bypasses VGPRs).
__device__ __forceinline__ void async_b128(unsigned short* lds,
                                           const unsigned short* g) {
    unsigned off = (unsigned)(size_t)lds;   // low 32 bits = LDS offset
    asm volatile("global_load_async_to_lds_b128 %0, %1, off"
                 :: "v"(off), "v"(g) : "memory");
}
#define WAIT_ASYNC(n) asm volatile("s_wait_asynccnt " #n ::: "memory")

// ---------------------------------------------------------------------------
// Elementwise convert kernels (fp32 -> bf16), plain and transposed
// ---------------------------------------------------------------------------
__global__ __launch_bounds__(256) void cvt_kernel(const float* __restrict__ in,
                                                  unsigned short* __restrict__ out,
                                                  int n) {
    int i = blockIdx.x * 256 + threadIdx.x;
    if (i < n) out[i] = f2bf(in[i]);
}

__global__ __launch_bounds__(256) void cvtT_kernel(const float* __restrict__ in,
                                                   unsigned short* __restrict__ out,
                                                   int rows, int cols) {
    int i = blockIdx.x * 256 + threadIdx.x;
    if (i < rows * cols) {
        int r = i / cols, c = i % cols;
        out[c * rows + r] = f2bf(in[i]);   // out = in^T
    }
}

// ---------------------------------------------------------------------------
// Tiled bf16 WMMA GEMM:  C[ROWS x 768] = A[ROWS x 768] * W  (+bias ...)
// A: bf16 row-major.  WT: bf16, W transposed -> WT[n][k] row-major.
// MODE 0: out bf16, head layout       [B,H,S,DK]
// MODE 1: out bf16, head-transposed   [B,H,DK,S]   (for V)
// MODE 2: out f32 = acc + bias + residual, flat [ROWS,768]
// 256 threads (8 waves), tile 128x128, K-step 32.
// Triple-buffered async global->LDS pipeline: ONE barrier per K-step,
// copies of tile i+2 overlap WMMA of tile i.
// ---------------------------------------------------------------------------
#define GA_STRIDE 40                    // LDS row stride (elems), 80 B
#define GA_TILE   (128 * GA_STRIDE)

template <int MODE>
__global__ __launch_bounds__(256, 2)
void gemm_kernel(const unsigned short* __restrict__ A,
                 const unsigned short* __restrict__ WT,
                 const float* __restrict__ bias,
                 const float* __restrict__ resid,
                 void* __restrict__ outp) {
    __shared__ unsigned short Alds[3 * GA_TILE];
    __shared__ unsigned short Blds[3 * GA_TILE];

    const int t      = threadIdx.x;
    const int wave   = t >> 5;
    const int lane   = t & 31;
    const int lane16 = lane & 15;
    const int half   = lane >> 4;
    const int kbase  = half * 8;
    const int wm     = wave >> 1;    // 0..3
    const int wn     = wave & 1;     // 0..1

    const int blockN = blockIdx.x * 128;
    const int blockM = blockIdx.y * 128;

    v8f acc[2][4];
#pragma unroll
    for (int a = 0; a < 2; ++a)
#pragma unroll
        for (int b = 0; b < 4; ++b)
#pragma unroll
            for (int j = 0; j < 8; ++j) acc[a][b][j] = 0.f;

    const int lrow = t >> 2;          // 0..63
    const int lcol = (t & 3) * 8;     // 0,8,16,24

    // one 128x32 A tile + 128x32 B tile = 4 async instructions per wave
    auto issue_tile = [&](int buf, int k0) {
#pragma unroll
        for (int p = 0; p < 2; ++p) {
            int r = lrow + 64 * p;
            async_b128(&Alds[buf * GA_TILE + r * GA_STRIDE + lcol],
                       &A[(size_t)(blockM + r) * D_MODEL + k0 + lcol]);
            async_b128(&Blds[buf * GA_TILE + r * GA_STRIDE + lcol],
                       &WT[(size_t)(blockN + r) * D_MODEL + k0 + lcol]);
        }
    };

    const int NK = D_MODEL / 32;       // 24 K-steps
    issue_tile(0, 0);
    issue_tile(1, 32);

    for (int kk = 0; kk < NK; ++kk) {
        const int bc = kk % 3;
        // tile kk done (tile kk+1's 4 asyncs may still be in flight)
        if (kk + 1 < NK) { WAIT_ASYNC(4); } else { WAIT_ASYNC(0); }
        // barrier: all waves have tile kk visible AND have finished
        // computing tile kk-1 -> buffer (kk+2)%3 is reusable.
        __syncthreads();
        if (kk + 2 < NK) issue_tile((bc + 2) % 3, (kk + 2) * 32);

        Frag16 af[2], bf[4];
#pragma unroll
        for (int mt = 0; mt < 2; ++mt)
            load_frag(af[mt],
                      &Alds[bc * GA_TILE + (wm * 32 + mt * 16 + lane16) * GA_STRIDE],
                      kbase);
#pragma unroll
        for (int nt = 0; nt < 4; ++nt)
            load_frag(bf[nt],
                      &Blds[bc * GA_TILE + (wn * 64 + nt * 16 + lane16) * GA_STRIDE],
                      kbase);
#pragma unroll
        for (int mt = 0; mt < 2; ++mt)
#pragma unroll
            for (int nt = 0; nt < 4; ++nt)
                acc[mt][nt] = __builtin_amdgcn_wmma_f32_16x16x32_bf16(
                    false, af[mt].v, false, bf[nt].v, (short)0, acc[mt][nt],
                    false, false);
    }

    // Store
#pragma unroll
    for (int mt = 0; mt < 2; ++mt)
#pragma unroll
        for (int nt = 0; nt < 4; ++nt)
#pragma unroll
            for (int j = 0; j < 8; ++j) {
                int r = blockM + wm * 32 + mt * 16 + j + half * 8;
                int c = blockN + wn * 64 + nt * 16 + lane16;
                float val = acc[mt][nt][j] + bias[c];
                if constexpr (MODE == 2) {
                    ((float*)outp)[(size_t)r * D_MODEL + c] =
                        val + resid[(size_t)r * D_MODEL + c];
                } else {
                    int bi = r >> 11, s = r & 2047;
                    int h = c >> 6, d = c & 63;
                    size_t idx;
                    if constexpr (MODE == 0)
                        idx = (((size_t)bi * N_HEADS + h) * SEQ + s) * D_K + d;
                    else
                        idx = (((size_t)bi * N_HEADS + h) * D_K + d) * SEQ + s;
                    ((unsigned short*)outp)[idx] = f2bf(val);
                }
            }
}

// ---------------------------------------------------------------------------
// Flash attention: one block per (64 query rows, head, batch).
// 4 waves x 16 rows. KV tiles of 64, triple-buffered async global->LDS
// staging shared by all 4 waves (one barrier per kv-step). Online softmax
// in registers; P routed C-layout -> A-layout through per-wave LDS tile.
// Qh, Kh: [B,H,S,DK] bf16.  Vt: [B,H,DK,S] bf16.  Out: [ROWS, 768] bf16.
// ---------------------------------------------------------------------------
#define AT_STRIDE 72                   // LDS row stride (elems), 144 B
#define AT_TILE   (64 * AT_STRIDE)

__global__ __launch_bounds__(128, 2)
void attn_kernel(const unsigned short* __restrict__ Qh,
                 const unsigned short* __restrict__ Kh,
                 const unsigned short* __restrict__ Vt,
                 unsigned short* __restrict__ Oout) {
    __shared__ unsigned short Klds[3 * AT_TILE];   // K tile  [kv][d]
    __shared__ unsigned short Vlds[3 * AT_TILE];   // Vt tile [d][kv]
    __shared__ unsigned short Plds[4][16 * AT_STRIDE];

    const int t      = threadIdx.x;
    const int wave   = t >> 5;
    const int lane   = t & 31;
    const int lane16 = lane & 15;
    const int half   = lane >> 4;
    const int kbase  = half * 8;

    const int b  = blockIdx.z;
    const int h  = blockIdx.y;
    const int q0 = blockIdx.x * 64;

    const size_t hb      = (size_t)b * N_HEADS + h;
    const size_t qk_base = hb * SEQ * D_K;     // Qh / Kh head base
    const size_t vt_base = hb * D_K * SEQ;     // Vt head base

    // cooperative KV tile copy: 8 async b128 per wave per tile
    auto issue_kv = [&](int buf, int kv0) {
        int row = t >> 1;                 // 0..63
        int cb  = (t & 1) * 32;
#pragma unroll
        for (int c = 0; c < 4; ++c)
            async_b128(&Klds[buf * AT_TILE + row * AT_STRIDE + cb + c * 8],
                       &Kh[qk_base + (size_t)(kv0 + row) * D_K + cb + c * 8]);
#pragma unroll
        for (int c = 0; c < 4; ++c)
            async_b128(&Vlds[buf * AT_TILE + row * AT_STRIDE + cb + c * 8],
                       &Vt[vt_base + (size_t)row * SEQ + kv0 + cb + c * 8]);
    };

    // Q fragments (loop invariant): 16 rows x 64 (two K-steps of 32)
    Frag16 aQ[2];
    {
        const unsigned short* qp = &Qh[qk_base + (size_t)(q0 + wave * 16 + lane16) * D_K];
        load_frag(aQ[0], qp, kbase);
        load_frag(aQ[1], qp + 32, kbase);
    }

    v8f accO[4];
#pragma unroll
    for (int tt = 0; tt < 4; ++tt)
#pragma unroll
        for (int j = 0; j < 8; ++j) accO[tt][j] = 0.f;
    float rmax[8], rsum[8];
#pragma unroll
    for (int j = 0; j < 8; ++j) { rmax[j] = -1e30f; rsum[j] = 0.f; }

    issue_kv(0, 0);
    issue_kv(1, 64);

    for (int kv0 = 0; kv0 < SEQ; kv0 += 64) {
        const int bc = (kv0 >> 6) % 3;
        if (kv0 + 64 < SEQ) { WAIT_ASYNC(8); } else { WAIT_ASYNC(0); }
        __syncthreads();
        if (kv0 + 128 < SEQ) issue_kv((bc + 2) % 3, kv0 + 128);

        // scores: 16 x 64 tile  (K fragments from LDS)
        v8f sc[4];
#pragma unroll
        for (int tt = 0; tt < 4; ++tt) {
            const unsigned short* kp =
                &Klds[bc * AT_TILE + (tt * 16 + lane16) * AT_STRIDE];
            Frag16 bk0, bk1;
            load_frag(bk0, kp, kbase);
            load_frag(bk1, kp + 32, kbase);
            v8f z;
#pragma unroll
            for (int j = 0; j < 8; ++j) z[j] = 0.f;
            z = __builtin_amdgcn_wmma_f32_16x16x32_bf16(false, aQ[0].v, false, bk0.v,
                                                        (short)0, z, false, false);
            sc[tt] = __builtin_amdgcn_wmma_f32_16x16x32_bf16(false, aQ[1].v, false, bk1.v,
                                                             (short)0, z, false, false);
        }

        // online softmax (scale 1/sqrt(64) = 0.125)
#pragma unroll
        for (int j = 0; j < 8; ++j) {
            float s0 = sc[0][j] * 0.125f, s1 = sc[1][j] * 0.125f;
            float s2 = sc[2][j] * 0.125f, s3 = sc[3][j] * 0.125f;
            sc[0][j] = s0; sc[1][j] = s1; sc[2][j] = s2; sc[3][j] = s3;
            float mx = fmaxf(fmaxf(s0, s1), fmaxf(s2, s3));
#pragma unroll
            for (int m = 8; m >= 1; m >>= 1)
                mx = fmaxf(mx, __shfl_xor(mx, m, 32));
            float mnew  = fmaxf(rmax[j], mx);
            float alpha = exp2f((rmax[j] - mnew) * LOG2E);
            rmax[j] = mnew;
            float ps = 0.f;
#pragma unroll
            for (int tt = 0; tt < 4; ++tt) {
                float p = exp2f((sc[tt][j] - mnew) * LOG2E);
                sc[tt][j] = p;
                ps += p;
            }
#pragma unroll
            for (int m = 8; m >= 1; m >>= 1)
                ps += __shfl_xor(ps, m, 32);
            rsum[j] = rsum[j] * alpha + ps;
#pragma unroll
            for (int tt = 0; tt < 4; ++tt) accO[tt][j] *= alpha;
        }

        // P: C-layout -> LDS -> A-layout bf16 fragments (per-wave region)
#pragma unroll
        for (int tt = 0; tt < 4; ++tt)
#pragma unroll
            for (int j = 0; j < 8; ++j) {
                int m = j + half * 8;
                Plds[wave][m * AT_STRIDE + tt * 16 + lane16] = f2bf(sc[tt][j]);
            }
        Frag16 aP[2];
        load_frag(aP[0], &Plds[wave][lane16 * AT_STRIDE], kbase);
        load_frag(aP[1], &Plds[wave][lane16 * AT_STRIDE + 32], kbase);

        // O += P * V   (V fragments from LDS, rows = head dims)
#pragma unroll
        for (int tt = 0; tt < 4; ++tt) {
            const unsigned short* vp =
                &Vlds[bc * AT_TILE + (tt * 16 + lane16) * AT_STRIDE];
            Frag16 bv0, bv1;
            load_frag(bv0, vp, kbase);
            load_frag(bv1, vp + 32, kbase);
            accO[tt] = __builtin_amdgcn_wmma_f32_16x16x32_bf16(false, aP[0].v, false, bv0.v,
                                                               (short)0, accO[tt], false, false);
            accO[tt] = __builtin_amdgcn_wmma_f32_16x16x32_bf16(false, aP[1].v, false, bv1.v,
                                                               (short)0, accO[tt], false, false);
        }
    }

    // normalize + store to [ROWS, 768] bf16 (merged-head layout)
#pragma unroll
    for (int tt = 0; tt < 4; ++tt)
#pragma unroll
        for (int j = 0; j < 8; ++j) {
            float o = accO[tt][j] / rsum[j];
            int row = b * SEQ + q0 + wave * 16 + j + half * 8;
            int col = h * D_K + tt * 16 + lane16;
            Oout[(size_t)row * D_MODEL + col] = f2bf(o);
        }
}

// ---------------------------------------------------------------------------
// LayerNorm: one wave per row of 768
// ---------------------------------------------------------------------------
__global__ __launch_bounds__(256)
void ln_kernel(const float* __restrict__ y,
               const float* __restrict__ gamma,
               const float* __restrict__ beta,
               float* __restrict__ out) {
    const int wave = threadIdx.x >> 5;
    const int lane = threadIdx.x & 31;
    const int row  = blockIdx.x * 8 + wave;
    const float* x = y + (size_t)row * D_MODEL;

    float s = 0.f, ss = 0.f;
    for (int i = lane; i < D_MODEL; i += 32) {
        float v = x[i];
        s += v; ss += v * v;
    }
#pragma unroll
    for (int m = 16; m >= 1; m >>= 1) {
        s  += __shfl_xor(s, m, 32);
        ss += __shfl_xor(ss, m, 32);
    }
    const float inv_n = 1.0f / (float)D_MODEL;
    float mean = s * inv_n;
    float var  = ss * inv_n - mean * mean;
    float inv  = rsqrtf(var + 1e-5f);
    float* o = out + (size_t)row * D_MODEL;
    for (int i = lane; i < D_MODEL; i += 32)
        o[i] = (x[i] - mean) * inv * gamma[i] + beta[i];
}

// ---------------------------------------------------------------------------
// Host-side orchestration
// ---------------------------------------------------------------------------
extern "C" void kernel_launch(void* const* d_in, const int* in_sizes, int n_in,
                              void* d_out, int out_size, void* d_ws, size_t ws_size,
                              hipStream_t stream) {
    const float* q   = (const float*)d_in[0];
    const float* k   = (const float*)d_in[1];
    const float* v   = (const float*)d_in[2];
    const float* W_q = (const float*)d_in[3];
    const float* b_q = (const float*)d_in[4];
    const float* W_k = (const float*)d_in[5];
    const float* b_k = (const float*)d_in[6];
    const float* W_v = (const float*)d_in[7];
    const float* b_v = (const float*)d_in[8];
    const float* W_o = (const float*)d_in[9];
    const float* b_o = (const float*)d_in[10];
    const float* ln_g = (const float*)d_in[11];
    const float* ln_b = (const float*)d_in[12];
    float* out = (float*)d_out;

    // workspace carve-up (256B aligned)
    char* ws = (char*)d_ws;
    size_t off = 0;
    auto alloc = [&](size_t bytes) -> char* {
        char* p = ws + off;
        off = (off + bytes + 255) & ~(size_t)255;
        return p;
    };
    const size_t act_bf = (size_t)ROWS * D_MODEL * 2;     // 6.29 MB
    const size_t w_bf   = (size_t)D_MODEL * D_MODEL * 2;  // 1.18 MB

    unsigned short* qbf = (unsigned short*)alloc(act_bf);
    unsigned short* kbf = (unsigned short*)alloc(act_bf);
    unsigned short* vbf = (unsigned short*)alloc(act_bf);
    unsigned short* WqT = (unsigned short*)alloc(w_bf);
    unsigned short* WkT = (unsigned short*)alloc(w_bf);
    unsigned short* WvT = (unsigned short*)alloc(w_bf);
    unsigned short* WoT = (unsigned short*)alloc(w_bf);
    unsigned short* Qh  = (unsigned short*)alloc(act_bf);
    unsigned short* Kh  = (unsigned short*)alloc(act_bf);
    unsigned short* Vt  = (unsigned short*)alloc(act_bf);
    unsigned short* AO  = (unsigned short*)alloc(act_bf);
    float*          Y   = (float*)alloc((size_t)ROWS * D_MODEL * 4);
    (void)ws_size; (void)n_in; (void)in_sizes; (void)out_size;

    const int n_act = ROWS * D_MODEL;           // 3,145,728
    // 1) precision conversion (one pass over all operands)
    cvt_kernel<<<(n_act + 255) / 256, 256, 0, stream>>>(q, qbf, n_act);
    cvt_kernel<<<(n_act + 255) / 256, 256, 0, stream>>>(k, kbf, n_act);
    cvt_kernel<<<(n_act + 255) / 256, 256, 0, stream>>>(v, vbf, n_act);
    const int n_w = D_MODEL * D_MODEL;
    cvtT_kernel<<<(n_w + 255) / 256, 256, 0, stream>>>(W_q, WqT, D_MODEL, D_MODEL);
    cvtT_kernel<<<(n_w + 255) / 256, 256, 0, stream>>>(W_k, WkT, D_MODEL, D_MODEL);
    cvtT_kernel<<<(n_w + 255) / 256, 256, 0, stream>>>(W_v, WvT, D_MODEL, D_MODEL);
    cvtT_kernel<<<(n_w + 255) / 256, 256, 0, stream>>>(W_o, WoT, D_MODEL, D_MODEL);

    // 2) Q/K/V projections (WMMA GEMM), head / head-transposed layouts
    dim3 ggrid(D_MODEL / 128, ROWS / 128);      // (6, 32)
    gemm_kernel<0><<<ggrid, 256, 0, stream>>>(qbf, WqT, b_q, nullptr, Qh);
    gemm_kernel<0><<<ggrid, 256, 0, stream>>>(kbf, WkT, b_k, nullptr, Kh);
    gemm_kernel<1><<<ggrid, 256, 0, stream>>>(vbf, WvT, b_v, nullptr, Vt);

    // 3) flash attention
    dim3 agrid(SEQ / 64, N_HEADS, BATCH);       // (32, 12, 2)
    attn_kernel<<<agrid, 128, 0, stream>>>(Qh, Kh, Vt, AO);

    // 4) output projection + bias + residual (f32 out)
    gemm_kernel<2><<<ggrid, 256, 0, stream>>>(AO, WoT, b_o, q, Y);

    // 5) LayerNorm
    ln_kernel<<<ROWS / 8, 256, 0, stream>>>(Y, ln_g, ln_b, out);
}
